// aug_module_69432441307746
// MI455X (gfx1250) — matrix operand order repaired and checked by reading, not compiled, not run on hardware
//
#include <hip/hip_runtime.h>
#include <math.h>

#define NN   6144
#define FIN  512
#define HH   64
#define EPSF 1.1920929e-07f
#define TEMPF 1.0f

typedef __attribute__((ext_vector_type(2)))  float  v2f;
typedef __attribute__((ext_vector_type(8)))  float  v8f;
typedef __attribute__((ext_vector_type(8)))  __bf16 v8bf;
typedef __attribute__((ext_vector_type(16))) __bf16 v16bf;
typedef int v4i __attribute__((vector_size(16)));

// ---------------------------------------------------------------------------
// CDNA5 async global->LDS copy (16B per lane), with synchronous fallback.
// Builtin signature (from hipcc diagnostic): arg0 = AS(1) v4i*, arg1 = LDS v4i*.
// ---------------------------------------------------------------------------
#if __has_builtin(__builtin_amdgcn_global_load_async_to_lds_b128)
#define HAVE_ASYNC_CP 1
#else
#define HAVE_ASYNC_CP 0
#endif

__device__ __forceinline__ void cp16_g2s(const void* g, void* s) {
#if HAVE_ASYNC_CP
  __builtin_amdgcn_global_load_async_to_lds_b128(
      (__attribute__((address_space(1))) v4i*)(g),
      (__attribute__((address_space(3))) v4i*)(s), 0, 0);
#else
  *(uint4*)s = *(const uint4*)g;
#endif
}

template <int N>
__device__ __forceinline__ void wait_cp() {
#if HAVE_ASYNC_CP
#if __has_builtin(__builtin_amdgcn_s_wait_asynccnt)
  __builtin_amdgcn_s_wait_asynccnt(N);
#else
  asm volatile("s_wait_asynccnt %0" ::"i"(N) : "memory");
#endif
#endif
}

// ---------------------------------------------------------------------------
// bf16 fragment gather: two 16B chunks -> v16bf
// ---------------------------------------------------------------------------
__device__ __forceinline__ v16bf ld_frag(const __bf16* p0, const __bf16* p1) {
  v8bf lo = *(const v8bf*)p0;
  v8bf hi = *(const v8bf*)p1;
  v16bf r;
#pragma unroll
  for (int i = 0; i < 8; ++i) { r[i] = lo[i]; r[i + 8] = hi[i]; }
  return r;
}

// ---------------------------------------------------------------------------
// Generic fp32 WMMA GEMM (single pass): C = op(A@B (+bias)(relu))
// One wave -> 16(M) x 64(N) strip, V_WMMA_F32_16X16X4_F32.
// Used for the small MLP GEMMs (K = 64).
// ---------------------------------------------------------------------------
template <bool RELU, bool BIAS>
__global__ void gemm_f32_wmma(const float* __restrict__ A,
                              const float* __restrict__ B,
                              const float* __restrict__ bias,
                              float* __restrict__ C,
                              int M, int K, int Nc) {
  const int lane   = threadIdx.x & 31;
  const int wave   = threadIdx.x >> 5;
  const int tilesN = Nc >> 6;
  const int gw     = blockIdx.x * 8 + wave;
  if (gw >= (M >> 4) * tilesN) return;
  const int tm   = gw / tilesN;
  const int tn   = gw % tilesN;
  const int r16  = lane & 15;
  const int half = lane >> 4;
  const int arow = tm * 16 + r16;

  v8f acc[4];
#pragma unroll
  for (int t = 0; t < 4; ++t)
#pragma unroll
    for (int r = 0; r < 8; ++r) acc[t][r] = 0.0f;

  for (int k0 = 0; k0 < K; k0 += 4) {
    const int ka = k0 + half * 2;
    v2f a = *(const v2f*)(A + arow * K + ka);
#pragma unroll
    for (int t = 0; t < 4; ++t) {
      const int col = tn * 64 + t * 16 + r16;
      v2f b;
      b.x = B[(ka)*Nc + col];
      b.y = B[(ka + 1) * Nc + col];
      acc[t] = __builtin_amdgcn_wmma_f32_16x16x4_f32(
          false, a, false, b, (short)0, acc[t], false, false);
    }
  }
#pragma unroll
  for (int t = 0; t < 4; ++t) {
    const int col = tn * 64 + t * 16 + r16;
    const float bi = BIAS ? bias[col] : 0.0f;
#pragma unroll
    for (int r = 0; r < 8; ++r) {
      float v = acc[t][r] + bi;
      if (RELU) v = fmaxf(v, 0.0f);
      C[(tm * 16 + half * 8 + r) * Nc + col] = v;
    }
  }
}

// ---------------------------------------------------------------------------
// Split-K fp32 WMMA GEMM: partials accumulated with global_atomic_add_f32.
// C must be pre-zeroed; bias/relu applied by a separate epilogue kernel.
// Spreads the big K=6144 adj stream across the whole device.
// ---------------------------------------------------------------------------
__global__ void gemm_f32_wmma_splitk(const float* __restrict__ A,
                                     const float* __restrict__ B,
                                     float* __restrict__ C,
                                     int M, int K, int Nc, int splitK) {
  const int lane   = threadIdx.x & 31;
  const int wave   = threadIdx.x >> 5;
  const int tilesN = Nc >> 6;
  const int gw     = blockIdx.x * 8 + wave;
  if (gw >= (M >> 4) * tilesN * splitK) return;
  const int tile = gw / splitK;
  const int seg  = gw % splitK;
  const int tm   = tile / tilesN;
  const int tn   = tile % tilesN;
  const int r16  = lane & 15;
  const int half = lane >> 4;
  const int arow = tm * 16 + r16;
  const int kLen = K / splitK;
  const int kBeg = seg * kLen;

  v8f acc[4];
#pragma unroll
  for (int t = 0; t < 4; ++t)
#pragma unroll
    for (int r = 0; r < 8; ++r) acc[t][r] = 0.0f;

  for (int k0 = kBeg; k0 < kBeg + kLen; k0 += 4) {
    const int ka = k0 + half * 2;
    v2f a = *(const v2f*)(A + arow * K + ka);
#pragma unroll
    for (int t = 0; t < 4; ++t) {
      const int col = tn * 64 + t * 16 + r16;
      v2f b;
      b.x = B[(ka)*Nc + col];
      b.y = B[(ka + 1) * Nc + col];
      acc[t] = __builtin_amdgcn_wmma_f32_16x16x4_f32(
          false, a, false, b, (short)0, acc[t], false, false);
    }
  }
#pragma unroll
  for (int t = 0; t < 4; ++t) {
    const int col = tn * 64 + t * 16 + r16;
#pragma unroll
    for (int r = 0; r < 8; ++r)
      unsafeAtomicAdd(&C[(tm * 16 + half * 8 + r) * Nc + col], acc[t][r]);
  }
}

__global__ void zero_f32(float* __restrict__ p, int count) {
  const int i = blockIdx.x * 256 + threadIdx.x;
  if (i < count) p[i] = 0.0f;
}

template <bool RELU>
__global__ void bias_act(float* __restrict__ C, const float* __restrict__ bias,
                         int count, int Nc) {
  const int i = blockIdx.x * 256 + threadIdx.x;
  if (i >= count) return;
  float v = C[i] + bias[i % Nc];
  if (RELU) v = fmaxf(v, 0.0f);
  C[i] = v;
}

// ---------------------------------------------------------------------------
// adj_logits = Z @ Z^T, bf16 WMMA, LDS-staged and double-buffered.
// Block = 256 threads (8 waves) -> 128x128 output tile; wave -> 32x64.
// K-chunk = 32 (one WMMA K-step); tiles staged via async global->LDS.
// LDS rows padded to 40 bf16 (20 dwords) -> conflict-free ds_load_b128.
// ---------------------------------------------------------------------------
#define BM  128
#define BK  32
#define BKP 40
#define KCH (FIN / BK)  // 16 chunks

__global__ void __launch_bounds__(256)
gemm_zzt_bf16(const __bf16* __restrict__ Zb, float* __restrict__ Cout) {
  __shared__ __align__(16) __bf16 As[2][BM][BKP];
  __shared__ __align__(16) __bf16 Bs[2][BM][BKP];

  const int tid  = threadIdx.x;
  const int lane = tid & 31;
  const int wave = tid >> 5;
  const int r16  = lane & 15;
  const int half = lane >> 4;
  const int wm   = (wave & 3) * 32;   // wave M offset in block tile
  const int wn   = (wave >> 2) * 64;  // wave N offset in block tile
  const int rowBase = blockIdx.y * BM;
  const int colBase = blockIdx.x * BM;

  const __bf16* gA = Zb + rowBase * FIN;  // A rows
  const __bf16* gB = Zb + colBase * FIN;  // B rows (columns of Z^T)

  v8f acc[2][4];
#pragma unroll
  for (int m = 0; m < 2; ++m)
#pragma unroll
    for (int t = 0; t < 4; ++t)
#pragma unroll
      for (int r = 0; r < 8; ++r) acc[m][t][r] = 0.0f;

  // stage one K-chunk (128 rows x 32 bf16 each for A and B): 2+2 chunks/thread
  auto stage = [&](int nb, int ch) {
    const int kofs = ch * BK;
#pragma unroll
    for (int i = 0; i < 2; ++i) {
      const int cc  = tid + i * 256;
      const int row = cc >> 2;
      const int col = (cc & 3) * 8;
      cp16_g2s(gA + row * FIN + kofs + col, &As[nb][row][col]);
    }
#pragma unroll
    for (int i = 0; i < 2; ++i) {
      const int cc  = tid + i * 256;
      const int row = cc >> 2;
      const int col = (cc & 3) * 8;
      cp16_g2s(gB + row * FIN + kofs + col, &Bs[nb][row][col]);
    }
  };

  stage(0, 0);
  int buf = 0;
  for (int c = 0; c < KCH; ++c) {
    if (c + 1 < KCH) {
      stage(buf ^ 1, c + 1);
      wait_cp<4>();  // current chunk's 4 per-thread copies complete
    } else {
      wait_cp<0>();
    }
    __syncthreads();

    v16bf a[2], b[4];
#pragma unroll
    for (int m = 0; m < 2; ++m)
      a[m] = ld_frag(&As[buf][wm + m * 16 + r16][half * 8],
                     &As[buf][wm + m * 16 + r16][16 + half * 8]);
#pragma unroll
    for (int t = 0; t < 4; ++t)
      b[t] = ld_frag(&Bs[buf][wn + t * 16 + r16][half * 8],
                     &Bs[buf][wn + t * 16 + r16][16 + half * 8]);
#pragma unroll
    for (int m = 0; m < 2; ++m)
#pragma unroll
      for (int t = 0; t < 4; ++t)
        acc[m][t] = __builtin_amdgcn_wmma_f32_16x16x32_bf16(
            false, a[m], false, b[t], (short)0, acc[m][t], false, false);

    __syncthreads();
    buf ^= 1;
  }

#pragma unroll
  for (int m = 0; m < 2; ++m)
#pragma unroll
    for (int t = 0; t < 4; ++t) {
      const int col = colBase + wn + t * 16 + r16;
#pragma unroll
      for (int r = 0; r < 8; ++r)
        Cout[(rowBase + wm + m * 16 + half * 8 + r) * NN + col] = acc[m][t][r];
    }
}

// ---------------------------------------------------------------------------
// Elementwise helpers
// ---------------------------------------------------------------------------
__global__ void f32_to_bf16_kernel(const float* __restrict__ in,
                                   __bf16* __restrict__ out, int count) {
  const int i = blockIdx.x * 256 + threadIdx.x;
  if (i < count) out[i] = (__bf16)in[i];
}

__device__ __forceinline__ float clamp_probs(float p) {
  return fminf(fmaxf(p, EPSF), 1.0f - EPSF);
}

__device__ __forceinline__ float st_hard(float raw_logit, float u) {
  const float p     = clamp_probs(1.0f / (1.0f + __expf(-raw_logit)));
  const float l     = __logf(p) - log1pf(-p);
  const float uu    = clamp_probs(u);
  const float noise = __logf(uu) - log1pf(-uu);
  const float soft  = clamp_probs(1.0f / (1.0f + __expf(-(l + noise) / TEMPF)));
  return rintf(soft);
}

__global__ void sample_upper(const float* __restrict__ logits,
                             const float* __restrict__ u,
                             float* __restrict__ s) {
  const int i = blockIdx.y;
  const int j = blockIdx.x * 256 + threadIdx.x;
  if (j < i) return;
  const int idx = i * NN + j;
  s[idx] = (i == j) ? 1.0f : st_hard(logits[idx], u[idx]);
}

__global__ void mirror_lower(float* __restrict__ s) {
  __shared__ float tile[32][33];
  const int tj = blockIdx.x;
  const int ti = blockIdx.y;
  if (ti < tj) return;
  const int c = threadIdx.x & 31;
  for (int r = threadIdx.x >> 5; r < 32; r += 8)
    tile[r][c] = s[(tj * 32 + r) * NN + ti * 32 + c];
  __syncthreads();
  for (int r = threadIdx.x >> 5; r < 32; r += 8) {
    const int i = ti * 32 + r, j = tj * 32 + c;
    if (i > j) s[i * NN + j] = tile[c][r];
  }
}

__global__ void rowsum_rsqrt(const float* __restrict__ s, float* __restrict__ d) {
  __shared__ float red[256];
  const int i = blockIdx.x;
  float sum = 0.0f;
  for (int j = threadIdx.x; j < NN; j += 256) sum += s[i * NN + j];
  red[threadIdx.x] = sum;
  __syncthreads();
  for (int off = 128; off > 0; off >>= 1) {
    if (threadIdx.x < off) red[threadIdx.x] += red[threadIdx.x + off];
    __syncthreads();
  }
  if (threadIdx.x == 0) d[i] = rsqrtf(red[0]);
}

__global__ void normalize_adj(float* __restrict__ s, const float* __restrict__ d) {
  const int i = blockIdx.y;
  const int j = blockIdx.x * 256 + threadIdx.x;
  s[i * NN + j] *= d[i] * d[j];
}

__global__ void mask_x_kernel(const float* __restrict__ x,
                              const float* __restrict__ P,
                              const float* __restrict__ u,
                              float* __restrict__ out, int count) {
  const int i = blockIdx.x * 256 + threadIdx.x;
  if (i < count) out[i] = x[i] * st_hard(P[i], u[i]);
}

// ---------------------------------------------------------------------------
static inline int gemm_blocks(int M, int Nc, int splitK = 1) {
  const int waves = (M / 16) * (Nc / 64) * splitK;
  return (waves + 7) / 8;
}

extern "C" void kernel_launch(void* const* d_in, const int* in_sizes, int n_in,
                              void* d_out, int out_size, void* d_ws, size_t ws_size,
                              hipStream_t stream) {
  const float* adj    = (const float*)d_in[0];
  const float* x      = (const float*)d_in[1];
  const float* u_adj  = (const float*)d_in[2];
  const float* u_mask = (const float*)d_in[3];
  const float* W_gcn  = (const float*)d_in[4];
  const float* b_gcn  = (const float*)d_in[5];
  const float* Wa1    = (const float*)d_in[6];
  const float* ba1    = (const float*)d_in[7];
  const float* Wa2    = (const float*)d_in[8];
  const float* ba2    = (const float*)d_in[9];
  const float* Wx1    = (const float*)d_in[10];
  const float* bx1    = (const float*)d_in[11];
  const float* Wx2    = (const float*)d_in[12];
  const float* bx2    = (const float*)d_in[13];

  // Output regions (return order: adj_norm, x_new, adj_logits)
  float* adj_norm   = (float*)d_out;
  float* x_new      = adj_norm + (size_t)NN * NN;
  float* adj_logits = x_new + (size_t)NN * FIN;

  // Workspace carve-up (~36 MB)
  float*  XW = (float*)d_ws;                      // [N,H]
  float*  h  = XW + NN * HH;                      // [N,H]
  float*  T  = h + NN * HH;                       // [N,H]
  float*  Z  = T + NN * HH;                       // [N,F]
  float*  P  = Z + (size_t)NN * FIN;              // [N,F]
  __bf16* Zb = (__bf16*)(P + (size_t)NN * FIN);   // [N,F] bf16
  float*  dv = (float*)(Zb + (size_t)NN * FIN);   // [N]

  const dim3 blk(256);

  // --- GCN encoder (split-K to fill the device) ---
  zero_f32<<<(NN * HH + 255) / 256, blk, 0, stream>>>(XW, NN * HH);
  zero_f32<<<(NN * HH + 255) / 256, blk, 0, stream>>>(h, NN * HH);
  gemm_f32_wmma_splitk<<<gemm_blocks(NN, HH, 4), blk, 0, stream>>>(
      x, W_gcn, XW, NN, FIN, HH, 4);
  gemm_f32_wmma_splitk<<<gemm_blocks(NN, HH, 16), blk, 0, stream>>>(
      adj, XW, h, NN, NN, HH, 16);
  bias_act<true><<<(NN * HH + 255) / 256, blk, 0, stream>>>(h, b_gcn, NN * HH, HH);

  // --- MLPA -> Z ---
  gemm_f32_wmma<true, true><<<gemm_blocks(NN, HH), blk, 0, stream>>>(
      h, Wa1, ba1, T, NN, HH, HH);
  gemm_f32_wmma<false, true><<<gemm_blocks(NN, FIN), blk, 0, stream>>>(
      T, Wa2, ba2, Z, NN, HH, FIN);

  // --- adj_logits = Z @ Z^T (bf16 WMMA, LDS double-buffered) ---
  f32_to_bf16_kernel<<<(NN * FIN + 255) / 256, blk, 0, stream>>>(Z, Zb, NN * FIN);
  gemm_zzt_bf16<<<dim3(NN / BM, NN / BM), blk, 0, stream>>>(Zb, adj_logits);

  // --- MLPX -> mask logits, x_new = x * hard(mask) ---
  gemm_f32_wmma<true, true><<<gemm_blocks(NN, HH), blk, 0, stream>>>(
      h, Wx1, bx1, T, NN, HH, HH);
  gemm_f32_wmma<false, true><<<gemm_blocks(NN, FIN), blk, 0, stream>>>(
      T, Wx2, bx2, P, NN, HH, FIN);
  mask_x_kernel<<<(NN * FIN + 255) / 256, blk, 0, stream>>>(
      x, P, u_mask, x_new, NN * FIN);

  // --- sample + symmetrize + normalize, in place in adj_norm ---
  sample_upper<<<dim3(NN / 256, NN), blk, 0, stream>>>(adj_logits, u_adj, adj_norm);
  mirror_lower<<<dim3(NN / 32, NN / 32), blk, 0, stream>>>(adj_norm);
  rowsum_rsqrt<<<dim3(NN), blk, 0, stream>>>(adj_norm, dv);
  normalize_adj<<<dim3(NN / 256, NN), blk, 0, stream>>>(adj_norm, dv);

  (void)in_sizes; (void)n_in; (void)out_size; (void)ws_size;
}
